// Model_14431090115020
// MI455X (gfx1250) — matrix operand order repaired and checked by reading, not compiled, not run on hardware
//
#include <hip/hip_runtime.h>
#include <cstdint>
#include <cstddef>

// ---------------- model constants ----------------
#define B_    16
#define LBL_  48
#define PRED_ 336
#define CIN_  7
#define L_    (LBL_ + PRED_)   // 384
#define DM_   512
#define E_    2
#define DIN_  1024
#define NS_   16
#define DCONV_ 4
#define DTR_  32
#define TF_   4
#define M_    (B_ * L_)        // 6144
#define EPS_  1e-5f

typedef __attribute__((ext_vector_type(16))) _Float16 v16h;
typedef __attribute__((ext_vector_type(8)))  float    v8f;
typedef __attribute__((ext_vector_type(8)))  unsigned v8u;
typedef __attribute__((ext_vector_type(4)))  unsigned v4u;

// ---------------- small elementwise kernels ----------------

// mean/std over first LBL steps, per (b, c)
__global__ void norm_stats_kernel(const float* __restrict__ xdec,
                                  float* __restrict__ meanv,
                                  float* __restrict__ stdv) {
    int i = blockIdx.x * blockDim.x + threadIdx.x;
    if (i >= B_ * CIN_) return;
    int b = i / CIN_, c = i % CIN_;
    float s = 0.f;
    for (int l = 0; l < LBL_; ++l) s += xdec[((size_t)b * L_ + l) * CIN_ + c];
    float m = s / (float)LBL_;
    float v = 0.f;
    for (int l = 0; l < LBL_; ++l) {
        float d = xdec[((size_t)b * L_ + l) * CIN_ + c] - m;
        v += d * d;
    }
    v /= (float)LBL_;
    meanv[i] = m;
    stdv[i]  = sqrtf(v + EPS_);
}

// xd = concat(normalized head, tail)
__global__ void build_xd_kernel(const float* __restrict__ xdec,
                                const float* __restrict__ meanv,
                                const float* __restrict__ stdv,
                                float* __restrict__ xd) {
    int i = blockIdx.x * blockDim.x + threadIdx.x;
    if (i >= B_ * L_ * CIN_) return;
    int c = i % CIN_;
    int l = (i / CIN_) % L_;
    int b = i / (CIN_ * L_);
    float v = xdec[i];
    if (l < LBL_) v = (v - meanv[b * CIN_ + c]) / stdv[b * CIN_ + c];
    xd[i] = v;
}

// token conv1d (circular pad 1, kernel 3 over CIN) + time-feature proj -> residual stream
__global__ void token_embed_kernel(const float* __restrict__ xd,
                                   const float* __restrict__ xmark,
                                   const float* __restrict__ tw,   // (DM, CIN, 3)
                                   const float* __restrict__ fw,   // (DM, TF)
                                   float* __restrict__ x) {
    int i = blockIdx.x * blockDim.x + threadIdx.x;
    if (i >= M_ * DM_) return;
    int d = i % DM_;
    int l = (i / DM_) % L_;
    int b = i / (DM_ * L_);
    float acc = 0.f;
    #pragma unroll
    for (int kk = 0; kk < 3; ++kk) {
        int t = l + kk - 1;
        t = (t < 0) ? (L_ - 1) : ((t >= L_) ? 0 : t);
        const float* xp = xd + ((size_t)b * L_ + t) * CIN_;
        const float* wp = tw + (size_t)d * (CIN_ * 3) + kk;
        #pragma unroll
        for (int c = 0; c < CIN_; ++c) acc += xp[c] * wp[c * 3];
    }
    const float* mk = xmark + ((size_t)b * L_ + l) * TF_;
    #pragma unroll
    for (int t = 0; t < TF_; ++t) acc += mk[t] * fw[d * TF_ + t];
    x[i] = acc;
}

// f32 -> f16 convert (weights + activations)
__global__ void cvt_f16_kernel(const float* __restrict__ src,
                               _Float16* __restrict__ dst, int n) {
    int i = blockIdx.x * blockDim.x + threadIdx.x;
    if (i < n) dst[i] = (_Float16)src[i];
}

// layernorm of one 512-wide row -> f16
__global__ void ln_f16_kernel(const float* __restrict__ x,
                              const float* __restrict__ w,
                              const float* __restrict__ b,
                              _Float16* __restrict__ out) {
    __shared__ float red[256];
    int row = blockIdx.x;
    int t = threadIdx.x;
    const float* xr = x + (size_t)row * DM_;
    float v0 = xr[t], v1 = xr[t + 256];
    red[t] = v0 + v1;
    __syncthreads();
    for (int s = 128; s > 0; s >>= 1) { if (t < s) red[t] += red[t + s]; __syncthreads(); }
    float mean = red[0] * (1.0f / DM_);
    __syncthreads();
    float d0 = v0 - mean, d1 = v1 - mean;
    red[t] = d0 * d0 + d1 * d1;
    __syncthreads();
    for (int s = 128; s > 0; s >>= 1) { if (t < s) red[t] += red[t + s]; __syncthreads(); }
    float rstd = rsqrtf(red[0] * (1.0f / DM_) + EPS_);
    _Float16* o = out + (size_t)row * DM_;
    o[t]       = (_Float16)(d0 * rstd * w[t] + b[t]);
    o[t + 256] = (_Float16)(d1 * rstd * w[t + 256] + b[t + 256]);
}

// depthwise causal conv (DCONV=4) + bias + SiLU; writes f32 (for scan) and f16 (for xproj GEMM)
__global__ void conv_silu_kernel(const float* __restrict__ xz,      // (M, 2*DIN); xin = [:, :DIN]
                                 const float* __restrict__ cw,      // (DIN, 4)
                                 const float* __restrict__ cb,      // (DIN)
                                 float* __restrict__ xc,
                                 _Float16* __restrict__ xc16) {
    int i = blockIdx.x * blockDim.x + threadIdx.x;
    if (i >= M_ * DIN_) return;
    int d = i % DIN_;
    int l = (i / DIN_) % L_;
    int b = i / (DIN_ * L_);
    float acc = cb[d];
    #pragma unroll
    for (int j = 0; j < DCONV_; ++j) {
        int t = l - (DCONV_ - 1) + j;
        if (t >= 0) acc += xz[((size_t)b * L_ + t) * (2 * DIN_) + d] * cw[d * DCONV_ + j];
    }
    float s = acc / (1.f + __expf(-acc));   // silu
    xc[i]   = s;
    xc16[i] = (_Float16)s;
}

// dt = softplus(dt_pre + bias), in place
__global__ void softplus_kernel(float* __restrict__ dt, const float* __restrict__ bias) {
    int i = blockIdx.x * blockDim.x + threadIdx.x;
    if (i >= M_ * DIN_) return;
    int d = i % DIN_;
    float v = dt[i] + bias[d];
    dt[i] = (v > 20.f) ? v : log1pf(__expf(v));
}

// selective scan: one thread per (b, d); h[16] in registers; fused D-skip + SiLU(z) gate; f16 out
__global__ void scan_kernel(const float* __restrict__ xc,    // (M, DIN) u
                            const float* __restrict__ dt,    // (M, DIN)
                            const float* __restrict__ dbc,   // (M, 64): [dt_r | Bm | Cm]
                            const float* __restrict__ xz,    // (M, 2*DIN): z = [:, DIN:]
                            const float* __restrict__ Alog,  // (DIN, NS)
                            const float* __restrict__ Dp,    // (DIN)
                            _Float16* __restrict__ y16) {    // (M, DIN)
    int i = blockIdx.x * blockDim.x + threadIdx.x;
    if (i >= B_ * DIN_) return;
    int b = i / DIN_, d = i % DIN_;
    float a[NS_], h[NS_];
    #pragma unroll
    for (int n = 0; n < NS_; ++n) { a[n] = -__expf(Alog[d * NS_ + n]); h[n] = 0.f; }
    float Dv = Dp[d];
    for (int l = 0; l < L_; ++l) {
        size_t base = (size_t)b * L_ + l;
        float dtv = dt[base * DIN_ + d];
        float u   = xc[base * DIN_ + d];
        float zv  = xz[base * (2 * DIN_) + DIN_ + d];
        const float* bc = dbc + base * 64;
        float du = dtv * u;
        float y = 0.f;
        #pragma unroll
        for (int n = 0; n < NS_; ++n) {
            h[n] = __expf(dtv * a[n]) * h[n] + du * bc[DTR_ + n];
            y += h[n] * bc[DTR_ + NS_ + n];
        }
        float gate = zv / (1.f + __expf(-zv));
        y16[base * DIN_ + d] = (_Float16)((y + u * Dv) * gate);
    }
}

// final LN + (512->7) projection + denorm, last PRED steps
__global__ void final_out_kernel(const float* __restrict__ x,
                                 const float* __restrict__ fw,
                                 const float* __restrict__ fb,
                                 const float* __restrict__ ow,    // (CIN, DM)
                                 const float* __restrict__ meanv,
                                 const float* __restrict__ stdv,
                                 float* __restrict__ out) {
    __shared__ float xn[DM_];
    __shared__ float red[256];
    int bidx = blockIdx.x;                 // b*PRED + lp
    int b = bidx / PRED_, lp = bidx % PRED_;
    int row = b * L_ + (L_ - PRED_) + lp;
    int t = threadIdx.x;
    const float* xr = x + (size_t)row * DM_;
    float v0 = xr[t], v1 = xr[t + 256];
    red[t] = v0 + v1;
    __syncthreads();
    for (int s = 128; s > 0; s >>= 1) { if (t < s) red[t] += red[t + s]; __syncthreads(); }
    float mean = red[0] * (1.0f / DM_);
    __syncthreads();
    float d0 = v0 - mean, d1 = v1 - mean;
    red[t] = d0 * d0 + d1 * d1;
    __syncthreads();
    for (int s = 128; s > 0; s >>= 1) { if (t < s) red[t] += red[t + s]; __syncthreads(); }
    float rstd = rsqrtf(red[0] * (1.0f / DM_) + EPS_);
    xn[t]       = d0 * rstd * fw[t] + fb[t];
    xn[t + 256] = d1 * rstd * fw[t + 256] + fb[t + 256];
    __syncthreads();
    for (int c = 0; c < CIN_; ++c) {
        float s = 0.f;
        for (int k = t; k < DM_; k += 256) s += xn[k] * ow[c * DM_ + k];
        red[t] = s;
        __syncthreads();
        for (int st = 128; st > 0; st >>= 1) { if (t < st) red[t] += red[t + st]; __syncthreads(); }
        if (t == 0)
            out[((size_t)b * PRED_ + lp) * CIN_ + c] =
                red[0] * stdv[b * CIN_ + c] + meanv[b * CIN_ + c];
        __syncthreads();
    }
}

// ---------------- WMMA GEMM:  C[M,N] (+)= A[M,K] @ W[N,K]^T ----------------
// block = 256 threads = 8 waves; block tile 256 x 64; wave tile 32 x 64
// (8 accumulators, 2 A fragments, each B fragment reused twice).
// B tile (64 cols x 32 K) is staged through LDS (double-buffered, padded
// stride for bank spread); next-tile global load overlaps current WMMAs.
template <bool ACC>
__global__ __launch_bounds__(256)
void gemm_wmma_kernel(const _Float16* __restrict__ A, int lda,
                      const _Float16* __restrict__ W, int ldw,
                      float* __restrict__ C, int ldc, int K) {
    constexpr int BSTR = 40;                       // padded LDS row stride (halves)
    __shared__ _Float16 Bs[2][64 * BSTR];          // 2 x 5 KB

    const int tid  = threadIdx.x;
    const int wave = tid >> 5;
    const int lane = tid & 31;
    const int g    = lane >> 4;                    // K half-group selector
    const int ln   = lane & 15;
    const int row0 = blockIdx.y * 256 + wave * 32;
    const int col0 = blockIdx.x * 64;

    // cooperative B staging: thread -> (row 0..63, 8-half chunk 0..3)
    const int br = tid >> 2;
    const int bc = (tid & 3) * 8;
    const _Float16* Wst = W + (size_t)(col0 + br) * ldw + bc;

    // preload K-tile 0
    *reinterpret_cast<v4u*>(&Bs[0][br * BSTR + bc]) =
        *reinterpret_cast<const v4u*>(Wst);
    __syncthreads();

    v8f acc[2][4] = {};
    int buf = 0;
    for (int k0 = 0; k0 < K; k0 += 32, buf ^= 1) {
        const bool more = (k0 + 32) < K;
        v4u nxt = {};
        if (more) nxt = *reinterpret_cast<const v4u*>(Wst + (k0 + 32));

        // two A fragments direct from global (16x32 f16 ISA layout):
        // halves [g*8, g*8+8) then [16+g*8, +8)
        v16h af[2];
        #pragma unroll
        for (int m = 0; m < 2; ++m) {
            const _Float16* Ap = A + (size_t)(row0 + m * 16 + ln) * lda + k0 + g * 8;
            v4u a0 = *reinterpret_cast<const v4u*>(Ap);
            v4u a1 = *reinterpret_cast<const v4u*>(Ap + 16);
            v8u au;
            au[0] = a0[0]; au[1] = a0[1]; au[2] = a0[2]; au[3] = a0[3];
            au[4] = a1[0]; au[5] = a1[1]; au[6] = a1[2]; au[7] = a1[3];
            af[m] = __builtin_bit_cast(v16h, au);
        }

        #pragma unroll
        for (int t = 0; t < 4; ++t) {
            // B fragment (32x16): lane = column, halves [g*8, g*8+8) then [16+g*8, +8)
            const _Float16* Bp = &Bs[buf][(t * 16 + ln) * BSTR + g * 8];
            v4u b0 = *reinterpret_cast<const v4u*>(Bp);
            v4u b1 = *reinterpret_cast<const v4u*>(Bp + 16);
            v8u bu;
            bu[0] = b0[0]; bu[1] = b0[1]; bu[2] = b0[2]; bu[3] = b0[3];
            bu[4] = b1[0]; bu[5] = b1[1]; bu[6] = b1[2]; bu[7] = b1[3];
            v16h bf = __builtin_bit_cast(v16h, bu);
            acc[0][t] = __builtin_amdgcn_wmma_f32_16x16x32_f16(
                false, af[0], false, bf, (short)0, acc[0][t], false, false);
            acc[1][t] = __builtin_amdgcn_wmma_f32_16x16x32_f16(
                false, af[1], false, bf, (short)0, acc[1][t], false, false);
        }

        if (more)
            *reinterpret_cast<v4u*>(&Bs[buf ^ 1][br * BSTR + bc]) = nxt;
        __syncthreads();
    }

    // C/D layout: VGPR r -> row = g*8 + r, col = ln (within each 16x16 tile)
    #pragma unroll
    for (int m = 0; m < 2; ++m) {
        #pragma unroll
        for (int t = 0; t < 4; ++t) {
            #pragma unroll
            for (int r = 0; r < 8; ++r) {
                size_t off = (size_t)(row0 + m * 16 + g * 8 + r) * ldc
                           + (col0 + t * 16 + ln);
                if (ACC) C[off] += acc[m][t][r];
                else     C[off]  = acc[m][t][r];
            }
        }
    }
}

static void launch_gemm(const _Float16* A, int lda, const _Float16* W, int ldw,
                        float* C, int ldc, int Mrows, int Ncols, int K,
                        bool accumulate, hipStream_t s) {
    dim3 grid(Ncols / 64, Mrows / 256);
    if (accumulate)
        gemm_wmma_kernel<true><<<grid, 256, 0, s>>>(A, lda, W, ldw, C, ldc, K);
    else
        gemm_wmma_kernel<false><<<grid, 256, 0, s>>>(A, lda, W, ldw, C, ldc, K);
}

static void launch_cvt(const float* src, _Float16* dst, int n, hipStream_t s) {
    cvt_f16_kernel<<<(n + 255) / 256, 256, 0, s>>>(src, dst, n);
}

// ---------------- orchestration ----------------
extern "C" void kernel_launch(void* const* d_in, const int* in_sizes, int n_in,
                              void* d_out, int out_size, void* d_ws, size_t ws_size,
                              hipStream_t stream) {
    (void)in_sizes; (void)n_in; (void)out_size; (void)ws_size;
    const float* x_dec    = (const float*)d_in[2];
    const float* x_mark   = (const float*)d_in[3];
    const float* token_w  = (const float*)d_in[4];
    const float* timef_w  = (const float*)d_in[5];
    const float* norm_w   = (const float*)d_in[6];
    const float* norm_b   = (const float*)d_in[7];
    const float* in_w     = (const float*)d_in[8];
    const float* conv_w   = (const float*)d_in[9];
    const float* conv_b   = (const float*)d_in[10];
    const float* xproj_w  = (const float*)d_in[11];
    const float* dtproj_w = (const float*)d_in[12];
    const float* dtproj_b = (const float*)d_in[13];
    const float* A_log    = (const float*)d_in[14];
    const float* Dp       = (const float*)d_in[15];
    const float* outp_w   = (const float*)d_in[16];
    const float* fin_w    = (const float*)d_in[17];
    const float* fin_b    = (const float*)d_in[18];
    const float* out_w    = (const float*)d_in[19];
    float* out = (float*)d_out;

    char* ws = (char*)d_ws;
    size_t off = 0;
    auto alloc = [&](size_t bytes) -> char* {
        char* p = ws + off;
        off += (bytes + 255) & ~(size_t)255;
        return p;
    };
    float*    meanv = (float*)alloc((size_t)B_ * CIN_ * 4);
    float*    stdv  = (float*)alloc((size_t)B_ * CIN_ * 4);
    float*    xd    = (float*)alloc((size_t)B_ * L_ * CIN_ * 4);
    float*    xres  = (float*)alloc((size_t)M_ * DM_ * 4);
    _Float16* a16   = (_Float16*)alloc((size_t)M_ * DIN_ * 2);   // reused activation slab
    float*    xz    = (float*)alloc((size_t)M_ * 2 * DIN_ * 4);
    float*    xc    = (float*)alloc((size_t)M_ * DIN_ * 4);
    float*    dbc   = (float*)alloc((size_t)M_ * 64 * 4);
    float*    dtb   = (float*)alloc((size_t)M_ * DIN_ * 4);
    _Float16* w16   = (_Float16*)alloc((size_t)2048 * 512 * 2);  // reused weight slab

    norm_stats_kernel<<<(B_ * CIN_ + 63) / 64, 64, 0, stream>>>(x_dec, meanv, stdv);
    build_xd_kernel<<<(B_ * L_ * CIN_ + 255) / 256, 256, 0, stream>>>(x_dec, meanv, stdv, xd);
    token_embed_kernel<<<(M_ * DM_ + 255) / 256, 256, 0, stream>>>(xd, x_mark, token_w, timef_w, xres);

    for (int e = 0; e < E_; ++e) {
        // x_ln (f16)
        ln_f16_kernel<<<M_, 256, 0, stream>>>(xres, norm_w + e * DM_, norm_b + e * DM_, a16);
        // xz = x_ln @ in_w^T   [6144 x 2048, K=512]
        launch_cvt(in_w + (size_t)e * 2048 * 512, w16, 2048 * 512, stream);
        launch_gemm(a16, DM_, w16, DM_, xz, 2 * DIN_, M_, 2 * DIN_, DM_, false, stream);
        // depthwise conv + silu
        conv_silu_kernel<<<(M_ * DIN_ + 255) / 256, 256, 0, stream>>>(
            xz, conv_w + (size_t)e * DIN_ * DCONV_, conv_b + e * DIN_, xc, a16);
        // dbc = xc @ xproj^T   [6144 x 64, K=1024]
        launch_cvt(xproj_w + (size_t)e * 64 * DIN_, w16, 64 * DIN_, stream);
        launch_gemm(a16, DIN_, w16, DIN_, dbc, 64, M_, 64, DIN_, false, stream);
        // dt_pre = dt_r @ dtproj^T  [6144 x 1024, K=32]  (A = f16(dbc), lda=64, first 32 cols)
        launch_cvt(dbc, a16, M_ * 64, stream);
        launch_cvt(dtproj_w + (size_t)e * DIN_ * DTR_, w16, DIN_ * DTR_, stream);
        launch_gemm(a16, 64, w16, DTR_, dtb, DIN_, M_, DIN_, DTR_, false, stream);
        softplus_kernel<<<(M_ * DIN_ + 255) / 256, 256, 0, stream>>>(dtb, dtproj_b + e * DIN_);
        // selective scan + D-skip + silu(z) gate -> f16
        scan_kernel<<<(B_ * DIN_ + 255) / 256, 256, 0, stream>>>(
            xc, dtb, dbc, xz, A_log + (size_t)e * DIN_ * NS_, Dp + e * DIN_, a16);
        // x += y @ outproj^T   [6144 x 512, K=1024], fused residual accumulate
        launch_cvt(outp_w + (size_t)e * DM_ * DIN_, w16, DM_ * DIN_, stream);
        launch_gemm(a16, DIN_, w16, DIN_, xres, DM_, M_, DM_, DIN_, true, stream);
    }

    final_out_kernel<<<B_ * PRED_, 256, 0, stream>>>(xres, fin_w, fin_b, out_w, meanv, stdv, out);
}
